// SelfMHA_85787676771081
// MI455X (gfx1250) — matrix operand order repaired and checked by reading, not compiled, not run on hardware
//
#include <hip/hip_runtime.h>

// ---------------------------------------------------------------------------
// MI455X (gfx1250) fused self-MHA:
//   3x GEMM (bf16 WMMA) -> flash attention (bf16 WMMA + TDM tile staging)
//   -> GEMM
// Problem: B=2, N=2048, DIM=1024, H=16, HD=64  (fp32 in/out)
// Workspace: Qh/Kh bf16 [B,H,N,64], Vt bf16 [B,H,64,N], AO bf16 [B,N,1024]
//            = 4 * 4Mi elems * 2B = 32 MB of d_ws.
//
// Roofline: ~70 GFLOP total; ~160 MB working set (fits 192 MB L2).
// bf16 WMMA (16x16x32) is the fast matrix path on gfx1250; fp32 stays in the
// accumulators. TDM stages shared K/V tiles into LDS (4 waves reuse them),
// tracked with TENSORcnt, so the VMEM pipe only carries per-wave Q/P traffic.
// ---------------------------------------------------------------------------

typedef __attribute__((ext_vector_type(16))) __bf16          bf16x16;
typedef __attribute__((ext_vector_type(8)))  float           f32x8;
typedef __attribute__((ext_vector_type(8)))  unsigned short  u16x8;

namespace {

constexpr int Bc   = 2;
constexpr int Nc   = 2048;
constexpr int DIMc = 1024;
constexpr int Hc   = 16;
constexpr int HDc  = 64;
constexpr int Mrows = Bc * Nc;          // 4096
constexpr float SCALEc = 0.125f;        // 64^-0.5

__device__ __forceinline__ unsigned short f2bf(float x) {
  union { __bf16 b; unsigned short u; } c;
  c.b = (__bf16)x;                      // native cvt on gfx1250
  return c.u;
}

// Assemble a 16-element bf16 fragment from two aligned 8-element runs.
__device__ __forceinline__ bf16x16 load_frag(const unsigned short* p0,
                                             const unsigned short* p1) {
  union { bf16x16 v; u16x8 h[2]; } u;
  u.h[0] = *(const u16x8*)p0;
  u.h[1] = *(const u16x8*)p1;
  return u.v;
}

__device__ __forceinline__ f32x8 wmma_bf16(bf16x16 a, bf16x16 b, f32x8 c) {
  return __builtin_amdgcn_wmma_f32_16x16x32_bf16(
      /*neg_a=*/false, a, /*neg_b=*/false, b,
      /*c_mod=*/(short)0, c, /*reuse_a=*/false, /*reuse_b=*/false);
}

} // namespace

// ---------------------------------------------------------------------------
// Tensor Data Mover: 2D tile Global -> LDS.  This toolchain exposes the
// 6-arg builtin: (u32x4 g0, i32x8 g1, i32x4 g2, i32x4 g3, i32x8 pad, i32 cpol).
// D# bitfields per CDNA5 ISA 8.3/8.4: group0 {count=1, lds_addr, global_addr,
// type=2}; group1 {data_size=2B, pad_enable, pad_interval, pad_amount,
// tensor_dim0/1, tile_dim0/1, tensor_dim0_stride}.  Guarded by __has_builtin
// with a cooperative-copy fallback so compilation never breaks.
// ---------------------------------------------------------------------------
#if __has_builtin(__builtin_amdgcn_tensor_load_to_lds) && defined(__AMDGCN__)
#define MHA_HAS_TDM 1
typedef __attribute__((ext_vector_type(4))) unsigned int u32x4;
typedef __attribute__((ext_vector_type(8))) int          i32x8;
typedef __attribute__((ext_vector_type(4))) int          i32x4;

__device__ __forceinline__ void tdm_load_2d(unsigned lds_off, const void* gaddr,
                                            unsigned dim0, unsigned dim1,
                                            unsigned tile0, unsigned tile1,
                                            unsigned long long stride0,
                                            unsigned pad_int_code,
                                            unsigned pad_amt_code) {
  const unsigned long long ga = (unsigned long long)gaddr;
  u32x4 g0;
  g0[0] = 1u;                                            // count=1, user mode
  g0[1] = lds_off;                                       // lds_addr (bytes)
  g0[2] = (unsigned)(ga & 0xffffffffu);                  // global_addr[31:0]
  g0[3] = (unsigned)((ga >> 32) & 0x01ffffffu) | (2u << 30);  // [56:32] | type=2
  i32x8 g1;
  g1[0] = (int)((1u << 16) |                             // data_size = 2B
                (1u << 20) |                             // pad_enable
                (pad_int_code << 22) | (pad_amt_code << 25));
  g1[1] = (int)((dim0 & 0xffffu) << 16);                 // tensor_dim0[15:0]
  g1[2] = (int)(((dim0 >> 16) & 0xffffu) | ((dim1 & 0xffffu) << 16));
  g1[3] = (int)(((dim1 >> 16) & 0xffffu) | ((tile0 & 0xffffu) << 16));
  g1[4] = (int)(tile1 & 0xffffu);                        // tile_dim1 (tile_dim2=0)
  g1[5] = (int)(stride0 & 0xffffffffu);                  // tensor_dim0_stride
  g1[6] = (int)((stride0 >> 32) & 0xffffu);
  g1[7] = 0;
  const i32x4 z4 = {0, 0, 0, 0};
  const i32x8 z8 = {0, 0, 0, 0, 0, 0, 0, 0};
  __builtin_amdgcn_tensor_load_to_lds(g0, g1, z4, z4, z8, 0);
}
#else
#define MHA_HAS_TDM 0
#endif

// ---------------------------------------------------------------------------
// GEMM: D = A (MxK) * W^T (W is N'xK row-major) + bias, K = N' = 1024, M = 4096
// ABF16: 0 = A is f32 (convert while staging), 1 = A is bf16 (ushort bits)
// EPI:   0 = bf16 head-split [B,H,N,HD]; 1 = f32 flat [M,DIM];
//        2 = bf16 head-split transposed [B,H,HD,N]  (V projection)
// Block: 256 threads (8 waves), tile 128x128; wave tile 32x64 (2x4 WMMA accs).
// ---------------------------------------------------------------------------
template<int ABF16, int EPI>
__global__ __launch_bounds__(256)
void mha_gemm_wmma(const void* __restrict__ Av,
                   const float* __restrict__ W,
                   const float* __restrict__ bias,
                   void* __restrict__ Dst)
{
  constexpr int K = DIMc;
  constexpr int LDA = 40;               // LDS row stride: 8-runs stay 16B aligned
  __shared__ unsigned short As[128 * LDA];
  __shared__ unsigned short Ws[128 * LDA];

  const int tid  = threadIdx.x;
  const int lane = tid & 31;
  const int wave = tid >> 5;
  const int wm   = wave & 3;            // 4 row groups of 32
  const int wn   = wave >> 2;           // 2 col groups of 64
  const int l15  = lane & 15;
  const int half = lane >> 4;
  const int rowBlk = blockIdx.y * 128;
  const int colBlk = blockIdx.x * 128;

  // Staging: both tiles are 128 rows x 32 k; 2 threads/row, 16 elems each.
  const int srow = tid >> 1;
  const int skh  = (tid & 1) * 16;

  f32x8 acc[2][4] = {};

  for (int k0 = 0; k0 < K; k0 += 32) {
    __syncthreads();                    // protect previous iteration's reads

    // ---- stage A tile ----
    if (ABF16) {
      const unsigned short* ap =
          (const unsigned short*)Av + (size_t)(rowBlk + srow) * K + k0 + skh;
      *(u16x8*)&As[srow * LDA + skh]     = *(const u16x8*)ap;
      *(u16x8*)&As[srow * LDA + skh + 8] = *(const u16x8*)(ap + 8);
      if (k0 + 32 < K) __builtin_prefetch(ap + 32, 0, 1);
    } else {
      const float* ap =
          (const float*)Av + (size_t)(rowBlk + srow) * K + k0 + skh;
      u16x8 lo, hi;
#pragma unroll
      for (int i = 0; i < 8; ++i) { lo[i] = f2bf(ap[i]); hi[i] = f2bf(ap[8 + i]); }
      *(u16x8*)&As[srow * LDA + skh]     = lo;
      *(u16x8*)&As[srow * LDA + skh + 8] = hi;
      if (k0 + 32 < K) __builtin_prefetch(ap + 32, 0, 1);
    }

    // ---- stage W tile ----
    {
      const float* wp = W + (size_t)(colBlk + srow) * K + k0 + skh;
      u16x8 lo, hi;
#pragma unroll
      for (int i = 0; i < 8; ++i) { lo[i] = f2bf(wp[i]); hi[i] = f2bf(wp[8 + i]); }
      *(u16x8*)&Ws[srow * LDA + skh]     = lo;
      *(u16x8*)&Ws[srow * LDA + skh + 8] = hi;
      if (k0 + 32 < K) __builtin_prefetch(wp + 32, 0, 1);
    }

    __syncthreads();

    // ---- fragment loads (ISA 7.12.2 layouts) + 8 WMMAs ----
    bf16x16 af[2], bfr[4];
#pragma unroll
    for (int mi = 0; mi < 2; ++mi) {
      const int r  = wm * 32 + mi * 16 + l15;   // A: row per lane
      const int kh = half * 8;
      af[mi] = load_frag(&As[r * LDA + kh], &As[r * LDA + 16 + kh]);
    }
#pragma unroll
    for (int ni = 0; ni < 4; ++ni) {
      const int c  = wn * 64 + ni * 16 + l15;   // B: N per lane
      const int kb = half * 8;
      bfr[ni] = load_frag(&Ws[c * LDA + kb], &Ws[c * LDA + kb + 16]);
    }
#pragma unroll
    for (int mi = 0; mi < 2; ++mi)
#pragma unroll
      for (int ni = 0; ni < 4; ++ni)
        acc[mi][ni] = wmma_bf16(af[mi], bfr[ni], acc[mi][ni]);
  }

  // ---- epilogue ----
#pragma unroll
  for (int mi = 0; mi < 2; ++mi) {
    const int rbase = rowBlk + wm * 32 + mi * 16 + half * 8;
#pragma unroll
    for (int ni = 0; ni < 4; ++ni) {
      const int j  = colBlk + wn * 64 + ni * 16 + l15;
      const float bj = bias[j];
#pragma unroll
      for (int r = 0; r < 8; ++r) {
        const int row = rbase + r;
        const float vv = acc[mi][ni][r] + bj;
        if (EPI == 1) {
          ((float*)Dst)[(size_t)row * DIMc + j] = vv;
        } else {
          const int b_ = row >> 11, n_ = row & (Nc - 1);
          const int h_ = j >> 6,   c_ = j & (HDc - 1);
          const unsigned short bvv = f2bf(vv);
          if (EPI == 0)   // [B,H,N,HD]
            ((unsigned short*)Dst)[(((size_t)(b_ * Hc + h_) * Nc + n_) << 6) + c_] = bvv;
          else            // EPI == 2: [B,H,HD,N]
            ((unsigned short*)Dst)[(((size_t)(b_ * Hc + h_) * HDc + c_) << 11) + n_] = bvv;
        }
      }
    }
  }
}

// ---------------------------------------------------------------------------
// Flash attention, causal. 4 waves/block, one 16-row query tile per wave.
// K/V tiles for each 32-key block are SHARED across the 4 waves: staged into
// LDS by the Tensor Data Mover (wave 0 issues, s_wait_tensorcnt, barrier).
// Fully-masked key blocks contribute exactly zero via the online softmax, so
// all waves run the same uniform kb loop (barrier-safe).
// ---------------------------------------------------------------------------
__global__ __launch_bounds__(128)
void mha_flash_attn(const unsigned short* __restrict__ Qh,
                    const unsigned short* __restrict__ Kh,
                    const unsigned short* __restrict__ Vt,
                    unsigned short* __restrict__ AO)
{
  constexpr int KLD = 72;   // K tile LDS row stride (64 + 8 pad) elems
  constexpr int VLD = 40;   // V tile LDS row stride (32 + 8 pad) elems
  constexpr int LDP = 40;
  __shared__ unsigned short Ks[32 * KLD];        // 32 keys x 64 c
  __shared__ unsigned short Vs[64 * VLD];        // 64 c    x 32 keys
  __shared__ unsigned short Plds[4 * 16 * LDP];

  const int tid  = threadIdx.x;
  const int lane = tid & 31;
  const int wave = tid >> 5;
  const int l15  = lane & 15;
  const int half = lane >> 4;
  const int bh   = blockIdx.y;                       // b*H + h
  const int q0   = blockIdx.x * 64 + wave * 16;

  const unsigned short* Qb = Qh + (size_t)bh * Nc * HDc;
  const unsigned short* Kb = Kh + (size_t)bh * Nc * HDc;
  const unsigned short* Vb = Vt + (size_t)bh * HDc * Nc;
  unsigned short* Pw = &Plds[wave * 16 * LDP];

  // Q fragments: A-layout, rows q0..q0+15, c split 0..31 / 32..63
  bf16x16 qf[2];
  {
    const unsigned short* qp = Qb + (size_t)(q0 + l15) * HDc + half * 8;
#pragma unroll
    for (int cc = 0; cc < 2; ++cc)
      qf[cc] = load_frag(qp + cc * 32, qp + cc * 32 + 16);
  }

  f32x8 o[4] = {};
  float m[8], l[8];
#pragma unroll
  for (int r = 0; r < 8; ++r) { m[r] = -__builtin_inff(); l[r] = 0.0f; }

  const int kendBlk = blockIdx.x * 64 + 64;          // block-max causal bound
  for (int kb = 0; kb < kendBlk; kb += 32) {
    __syncthreads();                                 // prior tile reads done

    // ---- stage shared K (32x64) and V^T (64x32) tiles into LDS ----
#if MHA_HAS_TDM
    if (wave == 0) {
      // K tile: 2D tensor [2048 x 64], tile 32 rows x 64; pad 4 DW per 32 DW
      tdm_load_2d((unsigned)(size_t)&Ks[0], Kb + (size_t)kb * HDc,
                  /*dim0=*/HDc, /*dim1=*/Nc, /*tile0=*/HDc, /*tile1=*/32,
                  /*stride0=*/HDc, /*pad_int=*/4, /*pad_amt=*/3);
      // V tile: 2D tensor [64 x 2048], tile 64 rows x 32; pad 4 DW per 16 DW
      tdm_load_2d((unsigned)(size_t)&Vs[0], Vb + kb,
                  /*dim0=*/Nc, /*dim1=*/HDc, /*tile0=*/32, /*tile1=*/HDc,
                  /*stride0=*/Nc, /*pad_int=*/3, /*pad_amt=*/3);
      __builtin_amdgcn_s_wait_tensorcnt(0);
    }
#else
    {
      // Cooperative fallback: 128 threads copy both tiles.
      const int krow = tid >> 2, kc = (tid & 3) * 16;           // 32x64
      const u16x8* kp = (const u16x8*)(Kb + (size_t)(kb + krow) * HDc + kc);
      *(u16x8*)&Ks[krow * KLD + kc]     = kp[0];
      *(u16x8*)&Ks[krow * KLD + kc + 8] = kp[1];
      const int vrow = tid >> 1, vc = (tid & 1) * 16;           // 64x32
      const u16x8* vp = (const u16x8*)(Vb + (size_t)vrow * Nc + kb + vc);
      *(u16x8*)&Vs[vrow * VLD + vc]     = vp[0];
      *(u16x8*)&Vs[vrow * VLD + vc + 8] = vp[1];
    }
#endif
    __syncthreads();

    // ---- S = Q K^T : two 16x16 tiles over this 32-key block ----
    f32x8 s[2];
#pragma unroll
    for (int ks = 0; ks < 2; ++ks) {
      const unsigned short* kp = &Ks[(ks * 16 + l15) * KLD + half * 8];
      f32x8 z = {};
      f32x8 t = wmma_bf16(qf[0], load_frag(kp,      kp + 16), z);
      s[ks]   = wmma_bf16(qf[1], load_frag(kp + 32, kp + 48), t);
    }

    // ---- scale + causal mask + running block max (C-layout) ----
    float mloc[8];
#pragma unroll
    for (int r = 0; r < 8; ++r) {
      const int qg = q0 + half * 8 + r;
      float s0 = s[0][r] * SCALEc;
      float s1 = s[1][r] * SCALEc;
      s0 = ((kb + l15)      > qg) ? -__builtin_inff() : s0;
      s1 = ((kb + 16 + l15) > qg) ? -__builtin_inff() : s1;
      s[0][r] = s0; s[1][r] = s1;
      mloc[r] = fmaxf(s0, s1);
    }
#pragma unroll
    for (int off = 1; off < 16; off <<= 1)
#pragma unroll
      for (int r = 0; r < 8; ++r)
        mloc[r] = fmaxf(mloc[r], __shfl_xor(mloc[r], off));

    float corr[8];
#pragma unroll
    for (int r = 0; r < 8; ++r) {
      const float mn = fmaxf(m[r], mloc[r]);
      corr[r] = __expf(m[r] - mn);
      m[r] = mn;
    }

    // ---- P = exp(S - m), row sums, stage P to LDS (C-layout write) ----
    float rsum[8];
#pragma unroll
    for (int r = 0; r < 8; ++r) {
      const float p0 = __expf(s[0][r] - m[r]);
      const float p1 = __expf(s[1][r] - m[r]);
      rsum[r] = p0 + p1;
      unsigned short* pr = Pw + (half * 8 + r) * LDP;
      pr[l15]      = f2bf(p0);
      pr[16 + l15] = f2bf(p1);
    }
#pragma unroll
    for (int off = 1; off < 16; off <<= 1)
#pragma unroll
      for (int r = 0; r < 8; ++r)
        rsum[r] += __shfl_xor(rsum[r], off);
#pragma unroll
    for (int r = 0; r < 8; ++r) l[r] = l[r] * corr[r] + rsum[r];

    // ---- rescale O by correction factors ----
#pragma unroll
    for (int ci = 0; ci < 4; ++ci)
#pragma unroll
      for (int r = 0; r < 8; ++r) o[ci][r] *= corr[r];

    // ---- reload P as A-fragment (same-wave DS ops are in order) ----
    const bf16x16 pf = load_frag(&Pw[l15 * LDP + half * 8],
                                 &Pw[l15 * LDP + 16 + half * 8]);

    // ---- O += P V : V^T tile in LDS -> contiguous B fragments ----
#pragma unroll
    for (int ci = 0; ci < 4; ++ci) {
      const unsigned short* vp = &Vs[(ci * 16 + l15) * VLD + half * 8];
      o[ci] = wmma_bf16(pf, load_frag(vp, vp + 16), o[ci]);
    }
  }

  // ---- epilogue: O /= l, write bf16 to AO[B,N,DIM] (heads merged) ----
  float inv[8];
#pragma unroll
  for (int r = 0; r < 8; ++r) inv[r] = 1.0f / l[r];
  const int b_ = bh >> 4;
  const int h_ = bh & 15;
#pragma unroll
  for (int ci = 0; ci < 4; ++ci) {
    const int col = h_ * HDc + ci * 16 + l15;
#pragma unroll
    for (int r = 0; r < 8; ++r) {
      const int qg = q0 + half * 8 + r;
      AO[((size_t)(b_ * Nc + qg)) * DIMc + col] = f2bf(o[ci][r] * inv[r]);
    }
  }
}

// ---------------------------------------------------------------------------
extern "C" void kernel_launch(void* const* d_in, const int* in_sizes, int n_in,
                              void* d_out, int out_size, void* d_ws, size_t ws_size,
                              hipStream_t stream) {
  (void)in_sizes; (void)n_in; (void)out_size; (void)ws_size;

  const float* q  = (const float*)d_in[0];
  const float* k  = (const float*)d_in[1];
  const float* v  = (const float*)d_in[2];
  // d_in[3]: causal mask (tril) — implemented analytically in the kernel.
  const float* Wq = (const float*)d_in[4];
  const float* bq = (const float*)d_in[5];
  const float* Wk = (const float*)d_in[6];
  const float* bk = (const float*)d_in[7];
  const float* Wv = (const float*)d_in[8];
  const float* bv = (const float*)d_in[9];
  const float* Wo = (const float*)d_in[10];
  const float* bo = (const float*)d_in[11];

  constexpr size_t TEN = (size_t)Bc * Nc * DIMc;   // 4Mi elements
  unsigned short* Qh = (unsigned short*)d_ws;
  unsigned short* Kh = Qh + TEN;
  unsigned short* Vt = Kh + TEN;
  unsigned short* AO = Vt + TEN;

  const dim3 gb(DIMc / 128, Mrows / 128);          // (8, 32)
  const dim3 tb(256);

  mha_gemm_wmma<0, 0><<<gb, tb, 0, stream>>>(q, Wq, bq, Qh);
  mha_gemm_wmma<0, 0><<<gb, tb, 0, stream>>>(k, Wk, bk, Kh);
  mha_gemm_wmma<0, 2><<<gb, tb, 0, stream>>>(v, Wv, bv, Vt);

  mha_flash_attn<<<dim3(Nc / 64, Bc * Hc), 128, 0, stream>>>(Qh, Kh, Vt, AO);

  mha_gemm_wmma<1, 1><<<gb, tb, 0, stream>>>(AO, Wo, bo, d_out);
}
